// LSTMModel_32590211842254
// MI455X (gfx1250) — compile-verified, matrix operand order checked by reading, not run on hardware
//
#include <hip/hip_runtime.h>
#include <math.h>

// ---------------- problem constants ----------------
#define B_   256
#define T_   512
#define D_   256
#define H_   512
#define KTOT 768            // D + H  (concat [x_t | h])
#define NK   (KTOT / 32)    // 24 K-steps
#define GHS  (4 * H_)       // 2048 gate rows

// ---------------- ws layout (bytes) ----------------
#define WS_WCAT  0                          // f16 [2048][768]  = 3,145,728 B
#define WS_BIAS  3145728                    // f32 [2048]
#define WS_H0    (WS_BIAS + 8192)           // f32 [256][512]
#define WS_H1    (WS_H0 + 524288)
#define WS_C     (WS_H1 + 524288)

typedef __attribute__((ext_vector_type(16))) _Float16 v16h;
typedef __attribute__((ext_vector_type(8)))  float    v8f;

struct __align__(16) U4 { unsigned int x, y, z, w; };
struct __align__(16) F4 { float x, y, z, w; };
union  Frag  { U4 q[2]; v16h v; };
union  Pack8 { _Float16 h[8]; U4 q; };

__device__ __forceinline__ float sigm(float v)  { return 1.f / (1.f + __expf(-v)); }
// branch-free tanh: saturates correctly via exp overflow/underflow
__device__ __forceinline__ float ftanh(float v) { return 2.f * sigm(2.f * v) - 1.f; }

// ============================================================
// prep: Wcat = f16([W_ih | W_hh]), bias = b_ih + b_hh, h0 = c = 0
// ============================================================
__global__ __launch_bounds__(256) void prep_kernel(
    const float* __restrict__ W_ih, const float* __restrict__ W_hh,
    const float* __restrict__ b_ih, const float* __restrict__ b_hh,
    _Float16* __restrict__ Wcat, float* __restrict__ bias,
    float* __restrict__ h0, float* __restrict__ c)
{
    const int idx0   = blockIdx.x * blockDim.x + threadIdx.x;
    const int stride = gridDim.x * blockDim.x;

    for (int i = idx0; i < GHS * KTOT; i += stride) {
        const int n = i / KTOT;
        const int k = i - n * KTOT;
        const float w = (k < D_) ? W_ih[n * D_ + k] : W_hh[n * H_ + (k - D_)];
        Wcat[i] = (_Float16)w;
    }
    for (int i = idx0; i < GHS; i += stride) bias[i] = b_ih[i] + b_hh[i];
    for (int i = idx0; i < B_ * H_; i += stride) { h0[i] = 0.f; c[i] = 0.f; }
}

// ============================================================
// one recurrence step (triple-buffered LDS, async weight staging
// two iterations ahead; steady-state wait = s_wait_asynccnt 2):
//   gates[B,4H] = [x_t | h_in] @ Wcat^T + bias  (f16 WMMA, f32 acc)
//   c = sig(f)*c + sig(i)*tanh(g);  h_out = sig(o)*tanh(c)
// grid = (B/64, H/32), block = 256 (8 wave32)
// ============================================================
__global__ __launch_bounds__(256) void lstm_step_kernel(
    const float* __restrict__ x, const _Float16* __restrict__ Wcat,
    const float* __restrict__ bias, const float* __restrict__ h_in,
    float* __restrict__ h_out, float* __restrict__ c, int t)
{
    // stride 40 halves (80 B): conflict-free 16-B fragment reads on 64 banks
    __shared__ __align__(16) _Float16 As[3][64 * 40];
    __shared__ __align__(16) _Float16 Bs[3][128 * 40];

    const int tid  = threadIdx.x;
    const int lane = tid & 31;
    const int wv   = tid >> 5;        // 0..7 (wave32)
    const int mt   = wv & 3;          // m-subtile
    const int js   = wv >> 2;         // j-subtile
    const int m0   = blockIdx.x * 64;
    const int j0   = blockIdx.y * 32;

    // A staging: thread -> (row 0..63, 8 k-cols); B: (n-row 0..127, 32B chunk)
    const int arow = tid >> 2;
    const int acb  = (tid & 3) * 8;
    const int brow = tid >> 1;
    const int bcb  = (tid & 1) * 16;
    const int bn   = (brow >> 5) * H_ + j0 + (brow & 31);   // global Wcat row

    const float*    xrow = x    + ((size_t)(m0 + arow) * T_ + t) * D_;
    const float*    hrow = h_in +  (size_t)(m0 + arow) * H_ - D_;  // index by kg
    const _Float16* wrow = Wcat + (size_t)bn * KTOT + bcb;

    v8f acc[4];
    #pragma unroll
    for (int g = 0; g < 4; ++g)
        acc[g] = (v8f){0.f,0.f,0.f,0.f,0.f,0.f,0.f,0.f};

    const int fr = lane & 15;          // fragment row (M for A, N for B)
    const int kb = (lane >> 4) * 8;    // fragment K chunk base

    // ---- staging helpers -------------------------------------------------
    auto stageA = [&](int buf, int kt) {
        const int kg = kt * 32 + acb;
        const float* src = (kg < D_) ? (xrow + kg) : (hrow + kg);
        F4 f0 = *(const F4*)(src);
        F4 f1 = *(const F4*)(src + 4);
        Pack8 p;
        p.h[0]=(_Float16)f0.x; p.h[1]=(_Float16)f0.y;
        p.h[2]=(_Float16)f0.z; p.h[3]=(_Float16)f0.w;
        p.h[4]=(_Float16)f1.x; p.h[5]=(_Float16)f1.y;
        p.h[6]=(_Float16)f1.z; p.h[7]=(_Float16)f1.w;
        *(U4*)&As[buf][arow * 40 + acb] = p.q;
    };
    auto stageB = [&](int buf, int kt) {
        // pure f16 global -> LDS copy on the CDNA5 async engine (ASYNCcnt).
        // the imm offset is applied to both the LDS and the global address.
        unsigned           lds = (unsigned)(size_t)&Bs[buf][brow * 40 + bcb];
        unsigned long long ga  = (unsigned long long)(size_t)(wrow + (size_t)kt * 32);
        asm volatile("global_load_async_to_lds_b128 %0, %1, off\n\t"
                     "global_load_async_to_lds_b128 %0, %1, off offset:16"
                     :: "v"(lds), "v"(ga) : "memory");
    };

    // ---- prologue: fill buffers 0 and 1 ---------------------------------
    stageB(0, 0);
    stageB(1, 1);
    stageA(0, 0);
    stageA(1, 1);
    // in-order ASYNCcnt: <=2 outstanding means stage(0)'s pair has landed
    asm volatile("s_wait_asynccnt 0x2" ::: "memory");
    __syncthreads();

    int b0 = 0, b1 = 1, b2 = 2;   // rotating buffer ids: consume b0, fill b2
    #pragma unroll 3
    for (int kt = 0; kt < NK; ++kt) {
        if (kt + 2 < NK) {
            stageB(b2, kt + 2);      // async first: engine starts immediately
            stageA(b2, kt + 2);
            if (kt + 4 < NK) {       // prefetch A source four steps ahead
                const int kg4 = (kt + 4) * 32 + acb;
                __builtin_prefetch((kg4 < D_) ? (xrow + kg4) : (hrow + kg4), 0, 1);
            }
        }

        // fragments per ISA 16-bit layout: lane holds K=[kb,kb+8) and +16
        Frag a;
        a.q[0] = *(const U4*)&As[b0][(mt * 16 + fr) * 40 + kb     ];
        a.q[1] = *(const U4*)&As[b0][(mt * 16 + fr) * 40 + kb + 16];

        Frag bf[4];
        #pragma unroll
        for (int g = 0; g < 4; ++g) {
            const int r = g * 32 + js * 16 + fr;
            bf[g].q[0] = *(const U4*)&Bs[b0][r * 40 + kb     ];
            bf[g].q[1] = *(const U4*)&Bs[b0][r * 40 + kb + 16];
        }
        #pragma unroll
        for (int g = 0; g < 4; ++g)
            acc[g] = __builtin_amdgcn_wmma_f32_16x16x32_f16(
                         false, a.v, false, bf[g].v, (short)0, acc[g], false, false);

        // steady state: only require the PREVIOUS iteration's staging (the
        // pair feeding the next consume) to be done; this iteration's pair
        // (<=2 outstanding, in-order) keeps flying through the barrier.
        if (kt + 2 < NK) asm volatile("s_wait_asynccnt 0x2" ::: "memory");
        else             asm volatile("s_wait_asynccnt 0x0" ::: "memory");
        __syncthreads();             // publish b2 / retire reads of b0

        const int tmp = b0; b0 = b1; b1 = b2; b2 = tmp;
    }

    // ---- pointwise LSTM update, all 4 gates in this wave's registers ----
    const int jg    = j0 + js * 16 + fr;
    const int mbase = m0 + mt * 16 + (lane >> 4) * 8;
    const float b_i = bias[           jg];
    const float b_f = bias[    H_  +  jg];
    const float b_g = bias[2 * H_  +  jg];
    const float b_o = bias[3 * H_  +  jg];

    #pragma unroll
    for (int r = 0; r < 8; ++r) {
        const size_t off = (size_t)(mbase + r) * H_ + jg;
        const float gi = acc[0][r] + b_i;
        const float gf = acc[1][r] + b_f;
        const float gg = acc[2][r] + b_g;
        const float go = acc[3][r] + b_o;
        const float cn = sigm(gf) * c[off] + sigm(gi) * ftanh(gg);
        c[off]     = cn;
        h_out[off] = sigm(go) * ftanh(cn);
    }
}

// ============================================================
// out[b] = sigmoid(h_last[b] . fc_w + fc_b)
// ============================================================
__global__ __launch_bounds__(64) void fc_kernel(
    const float* __restrict__ h, const float* __restrict__ fc_w,
    const float* __restrict__ fc_b, float* __restrict__ out)
{
    __shared__ float red[64];
    const int b = blockIdx.x;
    float p = 0.f;
    for (int j = threadIdx.x; j < H_; j += 64)
        p += h[(size_t)b * H_ + j] * fc_w[j];
    red[threadIdx.x] = p;
    __syncthreads();
    for (int s = 32; s > 0; s >>= 1) {
        if (threadIdx.x < s) red[threadIdx.x] += red[threadIdx.x + s];
        __syncthreads();
    }
    if (threadIdx.x == 0) out[b] = sigm(red[0] + fc_b[0]);
}

// ============================================================
extern "C" void kernel_launch(void* const* d_in, const int* in_sizes, int n_in,
                              void* d_out, int out_size, void* d_ws, size_t ws_size,
                              hipStream_t stream)
{
    const float* x    = (const float*)d_in[0];
    const float* W_ih = (const float*)d_in[1];
    const float* W_hh = (const float*)d_in[2];
    const float* b_ih = (const float*)d_in[3];
    const float* b_hh = (const float*)d_in[4];
    const float* fc_w = (const float*)d_in[5];
    const float* fc_b = (const float*)d_in[6];
    float* out = (float*)d_out;

    char* ws = (char*)d_ws;
    _Float16* Wcat = (_Float16*)(ws + WS_WCAT);
    float*    bias = (float*)   (ws + WS_BIAS);
    float*    h0   = (float*)   (ws + WS_H0);
    float*    h1   = (float*)   (ws + WS_H1);
    float*    c    = (float*)   (ws + WS_C);

    prep_kernel<<<1024, 256, 0, stream>>>(W_ih, W_hh, b_ih, b_hh, Wcat, bias, h0, c);

    float* hb[2] = { h0, h1 };
    for (int t = 0; t < T_; ++t) {
        lstm_step_kernel<<<dim3(B_ / 64, H_ / 32), 256, 0, stream>>>(
            x, Wcat, bias, hb[t & 1], hb[(t + 1) & 1], c, t);
    }
    // after t = 511, last h was written to hb[0]
    fc_kernel<<<B_, 64, 0, stream>>>(h0, fc_w, fc_b, out);
}
